// DecoderBlock_53395033424063
// MI455X (gfx1250) — matrix operand order repaired and checked by reading, not compiled, not run on hardware
//
#include <hip/hip_runtime.h>
#include <hip/hip_bf16.h>

// ---------------------------------------------------------------------------
// Decoder block for MI455X (gfx1250, wave32, WMMA).
// All GEMMs in bf16 -> f32 accumulate via v_wmma_f32_16x16x32_bf16.
// GEMM staging: A tile via GLOBAL_LOAD_ASYNC_TO_LDS (ASYNCcnt), W tile via
// Tensor Data Mover descriptor (TENSORcnt), with per-thread fallbacks.
// Attention softmax is over the HEAD axis (reference quirk) with a
// multiplicative causal mask applied before softmax only.
// ---------------------------------------------------------------------------

typedef __bf16 bf16;
typedef __attribute__((ext_vector_type(16))) bf16 v16bf;
typedef __attribute__((ext_vector_type(8)))  float v8f;

#define D_MODEL 1024
#define SEQ     2048
#define NB      2
#define NHEAD   16
#define HDIM    64
#define ROWS    (NB * SEQ)   // 4096

#if defined(__HIP_DEVICE_COMPILE__) && __has_builtin(__builtin_amdgcn_global_load_async_to_lds_b128)
#define HAS_ASYNC_LDS 1
#else
#define HAS_ASYNC_LDS 0
#endif

#if defined(__HIP_DEVICE_COMPILE__) && __has_builtin(__builtin_amdgcn_tensor_load_to_lds)
#define HAS_TDM 1
#else
#define HAS_TDM 0
#endif

#if HAS_TDM
#define ASYNC_INFLIGHT 4   // only the 4 A-tile DMAs per thread per stage
#else
#define ASYNC_INFLIGHT 6   // A (4) + W (2) per thread per stage
#endif

static __device__ __forceinline__ v8f vzero8() {
  v8f z = {0.f, 0.f, 0.f, 0.f, 0.f, 0.f, 0.f, 0.f};
  return z;
}

static __device__ __forceinline__ v8f wmma_bf16(v16bf a, v16bf b, v8f c) {
  return __builtin_amdgcn_wmma_f32_16x16x32_bf16(false, a, false, b,
                                                 (short)0, c, false, false);
}

static __device__ __forceinline__ float gelu_exact(float v) {
  return 0.5f * v * (1.0f + erff(v * 0.70710678118654752f));
}

#if HAS_ASYNC_LDS
// Builtin types its endpoints as b128 = v4i32 pointers (per hipcc diagnostic):
//   param0: global (AS1) int4* source, param1: LDS (AS3) int4* destination.
typedef int v4i_b128 __attribute__((vector_size(16)));
typedef __attribute__((address_space(1))) v4i_b128* gas_b128;
typedef __attribute__((address_space(3))) v4i_b128* las_b128;

// One b128 (8 x bf16) DMA from global to LDS, tracked by ASYNCcnt.
static __device__ __forceinline__ void async_b128(const bf16* gsrc, bf16* ldst) {
  __builtin_amdgcn_global_load_async_to_lds_b128((gas_b128)gsrc, (las_b128)ldst, 0, 0);
}
template <int N> static __device__ __forceinline__ void wait_async() {
#if __has_builtin(__builtin_amdgcn_s_wait_asynccnt)
  __builtin_amdgcn_s_wait_asynccnt(N);
#else
  asm volatile("s_wait_asynccnt %0" ::"i"(N) : "memory");
#endif
}
#endif

#if HAS_TDM
typedef unsigned int u32x4 __attribute__((ext_vector_type(4)));
typedef int i32x8 __attribute__((ext_vector_type(8)));
typedef int i32x4 __attribute__((ext_vector_type(4)));

template <int N> static __device__ __forceinline__ void wait_tensor() {
#if __has_builtin(__builtin_amdgcn_s_wait_tensorcnt)
  __builtin_amdgcn_s_wait_tensorcnt((short)N);
#else
  asm volatile("s_wait_tensorcnt %0" ::"i"(N) : "memory");
#endif
}

// LDS byte offset of a generic pointer that is known to point into LDS.
static __device__ __forceinline__ unsigned lds_off_u32(void* p) {
  return (unsigned)(unsigned long long)(__attribute__((address_space(3))) void*)p;
}

// TDM 2D tile load: rows x (rowDw dwords) tile from a row-major tensor with
// tensor_dim0_stride = strideDw dwords, into LDS at ldsOff with per-row
// padding (pad after every 32 dwords by 4 dwords -> 144B LDS row stride).
// D# layout per CDNA5 ISA ch. 10.8 / CDNA5 ISA.
// This toolchain exposes the 6-arg builtin (extra int32x8 descriptor group).
static __device__ __forceinline__ void tdm_load_2d(const bf16* gsrc, unsigned ldsOff,
                                                   unsigned rowDw, unsigned rows,
                                                   unsigned strideDw) {
  const unsigned long long ga = (unsigned long long)gsrc;
  u32x4 g0;
  g0[0] = 1u;                                   // count=1, user descriptor
  g0[1] = ldsOff;                               // lds_addr (bytes)
  g0[2] = (unsigned)(ga & 0xFFFFFFFFu);         // global_addr[31:0]
  g0[3] = (unsigned)((ga >> 32) & 0x1FFFFFFu) | (2u << 30);  // addr[56:32], type=2
  i32x8 g1;
  // workgroup_mask=0 (not in cluster), data_size=2 (4B), pad_enable=1,
  // pad_interval=4 (pad after 32 dwords), pad_amount=3 (4 dwords of pad).
  g1[0] = (int)((2u << 16) | (1u << 20) | (4u << 22) | (3u << 25));
  const unsigned td0 = strideDw;   // tensor_dim0 (full row length, dwords)
  const unsigned td1 = rows;       // tensor_dim1 (tile fits exactly)
  g1[1] = (int)((td0 & 0xFFFFu) << 16);                       // atomic_addr=0 | td0.lo
  g1[2] = (int)(((td0 >> 16) & 0xFFFFu) | ((td1 & 0xFFFFu) << 16));
  g1[3] = (int)(((td1 >> 16) & 0xFFFFu) | ((rowDw & 0xFFFFu) << 16));  // tile_dim0
  g1[4] = (int)(rows & 0xFFFFu);                              // tile_dim1 | tile_dim2=0
  g1[5] = (int)strideDw;                                      // tensor_dim0_stride[31:0]
  g1[6] = 0;                                                  // stride[47:32], dim1_stride.lo
  g1[7] = 0;
  const i32x4 z4 = {0, 0, 0, 0};                              // 2D: groups 2/3 unused
  const i32x8 z8 = {0, 0, 0, 0, 0, 0, 0, 0};                  // extra group (unused)
  __builtin_amdgcn_tensor_load_to_lds(g0, g1, z4, z4, z8, 0);
}
#endif

// A-matrix fragment (16x32 bf16, M x K) from row-major src (row stride ld).
// k order used here (same in load_b_nk): lane%16 = M row; VGPR i, half h:
//   K = (lane/16)*16 + 2i + h
static __device__ __forceinline__ v16bf load_a_rm(const bf16* src, int ld, int lane) {
  const int l = lane & 15, g = lane >> 4;
  const bf16* p = src + l * ld;
  v16bf a;
#pragma unroll
  for (int i = 0; i < 8; ++i) {
    const int kb = (g << 4) + (i << 1);
    a[2 * i]     = p[kb];
    a[2 * i + 1] = p[kb + 1];
  }
  return a;
}

// B-matrix fragment (32x16 bf16, K x N) where B = src^T and src is
// row-major [N][K] (weight-style). lane%16 = N; lanes 0-15: K=0-15,
// lanes 16-31: K=16-31; VGPR i half h: K = (lane/16)*16 + 2i + h.
static __device__ __forceinline__ v16bf load_b_nk(const bf16* src, int ld, int lane) {
  const int l = lane & 15, g = lane >> 4;
  const bf16* p = src + l * ld + (g << 4);
  v16bf b;
#pragma unroll
  for (int i = 0; i < 8; ++i) {
    b[2 * i]     = p[2 * i];
    b[2 * i + 1] = p[2 * i + 1];
  }
  return b;
}

// B-matrix fragment (32x16) from row-major [K][N] source (V-matrix style).
static __device__ __forceinline__ v16bf load_b_kn(const bf16* src, int ld, int lane) {
  const int l = lane & 15, g = lane >> 4;
  const bf16* p = src + (g << 4) * ld + l;
  v16bf b;
#pragma unroll
  for (int i = 0; i < 8; ++i) {
    b[2 * i]     = p[(2 * i) * ld];
    b[2 * i + 1] = p[(2 * i + 1) * ld];
  }
  return b;
}

// ---------------------------------------------------------------------------
// f32 -> bf16 convert (weights, once per launch; deterministic)
// ---------------------------------------------------------------------------
__global__ __launch_bounds__(256) void f32_to_bf16_kernel(const float* __restrict__ in,
                                                          bf16* __restrict__ out, int n) {
  int i = blockIdx.x * 256 + threadIdx.x;
  if (i < n) out[i] = (bf16)in[i];
}

// ---------------------------------------------------------------------------
// LayerNorm (ddof=1, eps added to std) -> bf16 output. One row per block.
// ---------------------------------------------------------------------------
__global__ __launch_bounds__(256) void ln_bf16_kernel(const float* __restrict__ x,
                                                      const float* __restrict__ alpha,
                                                      const float* __restrict__ beta,
                                                      bf16* __restrict__ y, int d) {
  __shared__ float red[256];
  const int row = blockIdx.x;
  const float* xr = x + (size_t)row * d;
  float s = 0.f, sq = 0.f;
  for (int i = threadIdx.x; i < d; i += 256) {
    float v = xr[i];
    s += v;
    sq += v * v;
  }
  red[threadIdx.x] = s;
  __syncthreads();
  for (int off = 128; off > 0; off >>= 1) {
    if ((int)threadIdx.x < off) red[threadIdx.x] += red[threadIdx.x + off];
    __syncthreads();
  }
  const float sum = red[0];
  __syncthreads();
  red[threadIdx.x] = sq;
  __syncthreads();
  for (int off = 128; off > 0; off >>= 1) {
    if ((int)threadIdx.x < off) red[threadIdx.x] += red[threadIdx.x + off];
    __syncthreads();
  }
  const float sumsq = red[0];
  const float mean = sum / (float)d;
  const float var = fmaxf((sumsq - sum * mean) / (float)(d - 1), 0.f);
  const float scale = alpha[0] / (sqrtf(var) + 1e-7f);
  const float bias = beta[0];
  bf16* yr = y + (size_t)row * d;
  for (int i = threadIdx.x; i < d; i += 256)
    yr[i] = (bf16)((xr[i] - mean) * scale + bias);
}

// ---------------------------------------------------------------------------
// Tiled GEMM: out[m][n] = act(A[m][:] . W[n][:] + bias[n]) * outscale (+resid)
// A: [M,K] bf16 row-major.  W: [N,K] bf16 row-major (computes A @ W^T).
// Block tile 128x64, BK=64, double-buffered LDS staging (A: async DMA,
// W: TDM descriptor), 8 waves, each wave 32x32 = 2x2 WMMA 16x16x32 tiles,
// 8 WMMAs per barrier pair.  qkv_mode: scatter bf16 to [B, H, S, 64].
// ---------------------------------------------------------------------------
#define LDA_S 72   // 64 + 8 pad elems: 144B row stride = 32 dwords + 4 pad dwords
__global__ __launch_bounds__(256) void gemm_kernel(const bf16* __restrict__ A,
                                                   const bf16* __restrict__ W,
                                                   const float* __restrict__ bias,
                                                   const float* __restrict__ resid,
                                                   float* __restrict__ outf,
                                                   bf16* __restrict__ outb, int M, int N,
                                                   int K, int act_gelu, int qkv_mode,
                                                   float outscale) {
  __shared__ bf16 As[2][128 * LDA_S];  // 2 x 18 KB
  __shared__ bf16 Ws[2][64 * LDA_S];   // 2 x 9 KB

  const int tid = threadIdx.x;
  const int lane = tid & 31;
  const int w = tid >> 5;
  const int wm = (w >> 1) << 5;  // wave M offset: 0,32,64,96
  const int wn = (w & 1) << 5;   // wave N offset: 0,32
  const int m0 = blockIdx.y * 128;
  const int n0 = blockIdx.x * 64;

  v8f c00 = vzero8(), c01 = vzero8(), c10 = vzero8(), c11 = vzero8();

  // Staging map: A tile 128x64 -> 4 b128/thread; W tile 64x64 -> 2 b128/thread
  const int ar = tid >> 1, ac = (tid & 1) << 5;  // A: row, 32-elem half
  const int wr = tid >> 2, wc = (tid & 3) << 4;  // W: row, 16-elem quarter

#if HAS_TDM
  const unsigned wsBase = lds_off_u32(&Ws[0][0]);
#endif

  auto stage = [&](int buf, int k0) {
    const bf16* gA = A + (size_t)(m0 + ar) * K + k0 + ac;
    bf16* lA = &As[buf][ar * LDA_S + ac];
#if HAS_ASYNC_LDS
    async_b128(gA, lA);
    async_b128(gA + 8, lA + 8);
    async_b128(gA + 16, lA + 16);
    async_b128(gA + 24, lA + 24);
#else
    {
      const uint4* g4 = reinterpret_cast<const uint4*>(gA);
      uint4* l4 = reinterpret_cast<uint4*>(lA);
      l4[0] = g4[0];
      l4[1] = g4[1];
      l4[2] = g4[2];
      l4[3] = g4[3];
      if (k0 + 64 < K) __builtin_prefetch(gA + 64, 0, 3);
    }
#endif
#if HAS_TDM
    // Wave 1 issues one TDM descriptor for the whole 64x64 W tile.
    if (w == 1)
      tdm_load_2d(W + (size_t)n0 * K + k0,
                  wsBase + (unsigned)buf * (64 * LDA_S * 2),
                  /*rowDw=*/32, /*rows=*/64, /*strideDw=*/(unsigned)(K >> 1));
#else
    {
      const bf16* gW = W + (size_t)(n0 + wr) * K + k0 + wc;
      bf16* lW = &Ws[buf][wr * LDA_S + wc];
#if HAS_ASYNC_LDS
      async_b128(gW, lW);
      async_b128(gW + 8, lW + 8);
#else
      reinterpret_cast<uint4*>(lW)[0] = reinterpret_cast<const uint4*>(gW)[0];
      reinterpret_cast<uint4*>(lW)[1] = reinterpret_cast<const uint4*>(gW)[1];
      if (k0 + 64 < K) __builtin_prefetch(gW + 64, 0, 3);
#endif
    }
#endif
  };

  stage(0, 0);  // prologue: fill buffer 0
  int buf = 0;
  for (int k0 = 0; k0 < K; k0 += 64) {
    const bool more = (k0 + 64) < K;
    if (more) stage(buf ^ 1, k0 + 64);  // DMA next tile into other buffer
#if HAS_ASYNC_LDS
    if (more) wait_async<ASYNC_INFLIGHT>();  // current done; next in flight
    else      wait_async<0>();
#endif
#if HAS_TDM
    if (w == 1) {
      if (more) wait_tensor<1>();  // current W tile done; next descriptor in flight
      else      wait_tensor<0>();
    }
#endif
    __syncthreads();
#pragma unroll
    for (int kk = 0; kk < 64; kk += 32) {
      v16bf a0 = load_a_rm(&As[buf][wm * LDA_S + kk], LDA_S, lane);
      v16bf a1 = load_a_rm(&As[buf][(wm + 16) * LDA_S + kk], LDA_S, lane);
      v16bf b0 = load_b_nk(&Ws[buf][wn * LDA_S + kk], LDA_S, lane);
      v16bf b1 = load_b_nk(&Ws[buf][(wn + 16) * LDA_S + kk], LDA_S, lane);
      c00 = wmma_bf16(a0, b0, c00);
      c01 = wmma_bf16(a0, b1, c01);
      c10 = wmma_bf16(a1, b0, c10);
      c11 = wmma_bf16(a1, b1, c11);
    }
    __syncthreads();
    buf ^= 1;
  }

  // Epilogue. C/D layout: element (m = r + 8*(lane/16), n = lane%16).
  const int l = lane & 15, g = lane >> 4;
  v8f ct[4] = {c00, c01, c10, c11};
#pragma unroll
  for (int t = 0; t < 4; ++t) {
    const int ti = t >> 1, tj = t & 1;
#pragma unroll
    for (int r = 0; r < 8; ++r) {
      const int mg = m0 + wm + ti * 16 + g * 8 + r;
      const int ng = n0 + wn + tj * 16 + l;
      float val = ct[t][r] + bias[ng];
      if (act_gelu) val = gelu_exact(val);
      val *= outscale;
      if (resid) val += resid[(size_t)mg * N + ng];
      if (qkv_mode) {
        const int bb = mg >> 11, ss = mg & 2047, hh = ng >> 6, dd = ng & 63;
        outb[(((size_t)(bb * NHEAD + hh) * SEQ) + ss) * HDIM + dd] = (bf16)val;
      } else {
        if (outf) outf[(size_t)mg * N + ng] = val;
        if (outb) outb[(size_t)mg * N + ng] = (bf16)val;
      }
    }
  }
}

// ---------------------------------------------------------------------------
// Attention with head-axis softmax (reference semantics):
//   denom[b,q,k] = sum_h exp(mask(q,k) * s[b,h,q,k]);  attn = exp(.)/denom
//   out[b,h,q,:] = sum over ALL k of attn * v           (mask NOT reapplied)
// Q already carries the 1/sqrt(64) scale. Block = (b, q-tile of 16),
// 512 threads = 16 waves. Phase 1: wave wv owns k-tiles wv, wv+16, ...
// and accumulates denom over heads in-order (deterministic, no atomics)
// into a 128 KB LDS table. Phase 2: wave wv = head, recomputes scores,
// normalizes, and does P @ V with WMMA via a per-wave LDS P tile.
// ---------------------------------------------------------------------------
__global__ __launch_bounds__(512) void attn_kernel(const bf16* __restrict__ Q,
                                                   const bf16* __restrict__ Kp,
                                                   const bf16* __restrict__ V,
                                                   bf16* __restrict__ O) {
  extern __shared__ char smem[];
  float* denom = reinterpret_cast<float*>(smem);               // [16][2048]
  bf16* Pst = reinterpret_cast<bf16*>(smem + 16 * SEQ * 4);    // [16][16][32]

  const int tid = threadIdx.x, lane = tid & 31, wv = tid >> 5;
  const int l = lane & 15, g = lane >> 4;
  const int b = blockIdx.y, qt = blockIdx.x;

  for (int i = tid; i < 16 * SEQ; i += 512) denom[i] = 0.f;
  __syncthreads();

  // -------- Phase 1: denominator over heads --------
  for (int h = 0; h < NHEAD; ++h) {
    const bf16* qp = Q + ((size_t)(b * NHEAD + h) * SEQ + qt * 16) * HDIM;
    const v16bf a0 = load_a_rm(qp, HDIM, lane);
    const v16bf a1 = load_a_rm(qp + 32, HDIM, lane);
    for (int kt = wv; kt < SEQ / 16; kt += 16) {  // wave owns disjoint k columns
      const bf16* kp = Kp + ((size_t)(b * NHEAD + h) * SEQ + kt * 16) * HDIM;
      v8f s = vzero8();
      s = wmma_bf16(a0, load_b_nk(kp, HDIM, lane), s);
      s = wmma_bf16(a1, load_b_nk(kp + 32, HDIM, lane), s);
      const int kg = kt * 16 + l;
#pragma unroll
      for (int r = 0; r < 8; ++r) {
        const int m = g * 8 + r;
        const float sc = (kg <= qt * 16 + m) ? s[r] : 0.f;  // multiplicative mask
        denom[m * SEQ + kg] += __expf(sc);                  // wave-exclusive slot
      }
    }
  }
  __syncthreads();

  // -------- Phase 2: normalize + P @ V (wave = head wv) --------
  const size_t hoff = (size_t)(b * NHEAD + wv) * SEQ * HDIM;
  const bf16* qp = Q + hoff + (size_t)qt * 16 * HDIM;
  const v16bf a0 = load_a_rm(qp, HDIM, lane);
  const v16bf a1 = load_a_rm(qp + 32, HDIM, lane);
  bf16* pw = Pst + wv * 16 * 32;
  v8f o0 = vzero8(), o1 = vzero8(), o2 = vzero8(), o3 = vzero8();

  for (int kt2 = 0; kt2 < SEQ / 32; ++kt2) {
#pragma unroll
    for (int half = 0; half < 2; ++half) {
      const int kt = kt2 * 2 + half;
      const bf16* kp = Kp + hoff + (size_t)kt * 16 * HDIM;
      v8f s = vzero8();
      s = wmma_bf16(a0, load_b_nk(kp, HDIM, lane), s);
      s = wmma_bf16(a1, load_b_nk(kp + 32, HDIM, lane), s);
      const int kg = kt * 16 + l;
#pragma unroll
      for (int r = 0; r < 8; ++r) {
        const int m = g * 8 + r;
        const float sc = (kg <= qt * 16 + m) ? s[r] : 0.f;
        const float p = __expf(sc) / denom[m * SEQ + kg];
        pw[m * 32 + half * 16 + l] = (bf16)p;  // stage P as A-matrix source
      }
    }
    const v16bf ap = load_a_rm(pw, 32, lane);  // wave-private LDS, no barrier
    const bf16* vp = V + hoff + (size_t)kt2 * 32 * HDIM;
    o0 = wmma_bf16(ap, load_b_kn(vp, HDIM, lane), o0);
    o1 = wmma_bf16(ap, load_b_kn(vp + 16, HDIM, lane), o1);
    o2 = wmma_bf16(ap, load_b_kn(vp + 32, HDIM, lane), o2);
    o3 = wmma_bf16(ap, load_b_kn(vp + 48, HDIM, lane), o3);
  }

  // Write [B*S, D] bf16 (proj-input layout), col = h*64 + d.
  v8f ot[4] = {o0, o1, o2, o3};
#pragma unroll
  for (int nc = 0; nc < 4; ++nc)
#pragma unroll
    for (int r = 0; r < 8; ++r) {
      const int m = g * 8 + r;
      const size_t row = (size_t)b * SEQ + qt * 16 + m;
      const int col = wv * HDIM + nc * 16 + l;
      O[row * D_MODEL + col] = (bf16)ot[nc][r];
    }
}

// ---------------------------------------------------------------------------
// Host orchestration
// ---------------------------------------------------------------------------
extern "C" void kernel_launch(void* const* d_in, const int* in_sizes, int n_in,
                              void* d_out, int out_size, void* d_ws, size_t ws_size,
                              hipStream_t stream) {
  (void)in_sizes; (void)n_in; (void)out_size; (void)ws_size;
  const float* x  = (const float*)d_in[0];
  // d_in[1] = mask (causal, multiplicative) -- computed analytically on device
  const float* Wq = (const float*)d_in[2];  const float* bq = (const float*)d_in[3];
  const float* Wk = (const float*)d_in[4];  const float* bk = (const float*)d_in[5];
  const float* Wv = (const float*)d_in[6];  const float* bv = (const float*)d_in[7];
  const float* Wo = (const float*)d_in[8];  const float* bo = (const float*)d_in[9];
  const float* W1 = (const float*)d_in[10]; const float* b1 = (const float*)d_in[11];
  const float* W2 = (const float*)d_in[12]; const float* b2 = (const float*)d_in[13];
  const float* a1 = (const float*)d_in[14]; const float* be1 = (const float*)d_in[15];
  const float* a2 = (const float*)d_in[16]; const float* be2 = (const float*)d_in[17];
  float* out = (float*)d_out;

  // Workspace carve-up (~84 MB total)
  size_t off = 0;
  char* wsb = (char*)d_ws;
  auto take = [&](size_t bytes) -> char* {
    char* p = wsb + off;
    off += (bytes + 255) & ~(size_t)255;
    return p;
  };
  const size_t WB = (size_t)D_MODEL * D_MODEL * sizeof(bf16);     // 2 MB
  const size_t AB = (size_t)ROWS * D_MODEL * sizeof(bf16);        // 8 MB
  bf16* Wqb = (bf16*)take(WB); bf16* Wkb = (bf16*)take(WB);
  bf16* Wvb = (bf16*)take(WB); bf16* Wob = (bf16*)take(WB);
  bf16* W1b = (bf16*)take(WB); bf16* W2b = (bf16*)take(WB);
  bf16* ybf  = (bf16*)take(AB);
  bf16* qb   = (bf16*)take(AB);
  bf16* kb   = (bf16*)take(AB);
  bf16* vb   = (bf16*)take(AB);
  bf16* ao   = (bf16*)take(AB);
  float* x1  = (float*)take((size_t)ROWS * D_MODEL * sizeof(float));  // 16 MB
  bf16* ybf2 = (bf16*)take(AB);
  bf16* hb   = (bf16*)take(AB);

  const int WN = D_MODEL * D_MODEL;  // 1M elems
  f32_to_bf16_kernel<<<WN / 256, 256, 0, stream>>>(Wq, Wqb, WN);
  f32_to_bf16_kernel<<<WN / 256, 256, 0, stream>>>(Wk, Wkb, WN);
  f32_to_bf16_kernel<<<WN / 256, 256, 0, stream>>>(Wv, Wvb, WN);
  f32_to_bf16_kernel<<<WN / 256, 256, 0, stream>>>(Wo, Wob, WN);
  f32_to_bf16_kernel<<<WN / 256, 256, 0, stream>>>(W1, W1b, WN);
  f32_to_bf16_kernel<<<WN / 256, 256, 0, stream>>>(W2, W2b, WN);

  // LN1
  ln_bf16_kernel<<<ROWS, 256, 0, stream>>>(x, a1, be1, ybf, D_MODEL);

  // Q/K/V projections (Q carries 1/sqrt(64) = 0.125 score scale)
  dim3 ggrid(D_MODEL / 64, ROWS / 128);
  gemm_kernel<<<ggrid, 256, 0, stream>>>(ybf, Wqb, bq, nullptr, nullptr, qb,
                                         ROWS, D_MODEL, D_MODEL, 0, 1, 0.125f);
  gemm_kernel<<<ggrid, 256, 0, stream>>>(ybf, Wkb, bk, nullptr, nullptr, kb,
                                         ROWS, D_MODEL, D_MODEL, 0, 1, 1.0f);
  gemm_kernel<<<ggrid, 256, 0, stream>>>(ybf, Wvb, bv, nullptr, nullptr, vb,
                                         ROWS, D_MODEL, D_MODEL, 0, 1, 1.0f);

  // Attention (head-axis softmax): 144 KB dynamic LDS per block (WGP has 320 KB)
  const size_t asmem = (size_t)16 * SEQ * 4 + (size_t)16 * 16 * 32 * 2;
  (void)hipFuncSetAttribute((const void*)attn_kernel,
                            hipFuncAttributeMaxDynamicSharedMemorySize, (int)asmem);
  attn_kernel<<<dim3(SEQ / 16, NB), 512, asmem, stream>>>(qb, kb, vb, ao);

  // Output projection + residual -> x1
  gemm_kernel<<<ggrid, 256, 0, stream>>>(ao, Wob, bo, x, x1, nullptr,
                                         ROWS, D_MODEL, D_MODEL, 0, 0, 1.0f);

  // LN2
  ln_bf16_kernel<<<ROWS, 256, 0, stream>>>(x1, a2, be2, ybf2, D_MODEL);

  // MLP: gelu(y@W1^T+b1) @ W2^T + b2 + residual -> out
  gemm_kernel<<<ggrid, 256, 0, stream>>>(ybf2, W1b, b1, nullptr, nullptr, hb,
                                         ROWS, D_MODEL, D_MODEL, 1, 0, 1.0f);
  gemm_kernel<<<ggrid, 256, 0, stream>>>(hb, W2b, b2, x1, out, nullptr,
                                         ROWS, D_MODEL, D_MODEL, 0, 0, 1.0f);
}
